// AdaptedBERTEncoderSingle_75582834475644
// MI455X (gfx1250) — matrix operand order repaired and checked by reading, not compiled, not run on hardware
//
#include <hip/hip_runtime.h>
#include <hip/hip_bf16.h>

// ---------------------------------------------------------------------------
// AdaptedBERTEncoder forward, MI455X / gfx1250, wave32 + WMMA bf16.
// Dense GEMMs *and* attention (QK^T, PV) run on v_wmma_f32_16x16x32_bf16.
// ---------------------------------------------------------------------------

#define B_  16
#define S_  128
#define P_  3
#define L_  386            // 1 + S*P + 1
#define D_  768
#define NL_ 4
#define H_  12
#define FF_ 3072
#define HD_ 64
#define M_  (B_ * L_)      // 6176 token rows
#define LP_ 416            // L_ padded to 13*32 for WMMA K-loop over kv

typedef __attribute__((ext_vector_type(16))) __bf16 v16bf;
typedef __attribute__((ext_vector_type(8)))  __bf16 v8bf;
typedef __attribute__((ext_vector_type(8)))  float  v8f;

union ABFrag { v16bf v; v8bf h[2]; };

// ---------------------------------------------------------------------------
// 1) old-vocab -> wordpiece remap + CLS/SEP + additive attention mask
// ---------------------------------------------------------------------------
__global__ void remap_kernel(const int* __restrict__ inpids,
                             const int* __restrict__ map_table,
                             const int* __restrict__ map_counts,
                             int* __restrict__ newinp,
                             float* __restrict__ maskbias)
{
  int bI = threadIdx.x;
  if (bI >= B_) return;
  int* row = newinp + bI * L_;
  for (int i = 0; i < L_; ++i) row[i] = 0;      // PAD
  row[0] = 101;                                 // CLS
  int off = 1, sep = 1;
  for (int s = 0; s < S_; ++s) {
    int id = inpids[bI * S_ + s];
    int c  = map_counts[id];
    if (c > P_) c = P_;
    for (int p = 0; p < c; ++p) {
      int pos = off + p;
      if (pos < L_) row[pos] = map_table[id * P_ + p];
    }
    off += c;
    if (id != 0) sep += c;
  }
  if (sep < L_) row[sep] = 102;                 // SEP
  for (int i = 0; i < L_; ++i)
    maskbias[bI * L_ + i] = (row[i] != 0) ? 0.0f : -1.0e4f;
}

// ---------------------------------------------------------------------------
// 2) f32 weights [mats][K][N]  ->  bf16 transposed [mats][N][K]
// ---------------------------------------------------------------------------
__global__ void convt_kernel(const float* __restrict__ src,
                             __bf16* __restrict__ dst,
                             int K, int N, long total)
{
  long e = (long)blockIdx.x * 256 + threadIdx.x;
  if (e >= total) return;
  long mn  = (long)K * N;
  long mat = e / mn;
  long rem = e - mat * mn;
  int  r   = (int)(rem / N);
  int  c   = (int)(rem - (long)r * N);
  dst[mat * mn + (long)c * K + r] = (__bf16)src[e];
}

// ---------------------------------------------------------------------------
// 3) embedding gather + LayerNorm, write bf16 activations
// ---------------------------------------------------------------------------
__global__ __launch_bounds__(256) void embed_ln_kernel(
    const int* __restrict__ newinp,
    const float* __restrict__ word_emb, const float* __restrict__ pos_emb,
    const float* __restrict__ type_emb,
    const float* __restrict__ g, const float* __restrict__ bta,
    __bf16* __restrict__ h)
{
  const int t = blockIdx.x;                 // token row 0..M_-1
  const int l = t % L_;
  const int tok = newinp[t];
  __shared__ float xs[D_];
  __shared__ float red[256];

  float local = 0.0f;
  for (int d = threadIdx.x; d < D_; d += 256) {
    float x = word_emb[(long)tok * D_ + d] + pos_emb[(long)l * D_ + d] + type_emb[d];
    xs[d] = x; local += x;
  }
  red[threadIdx.x] = local; __syncthreads();
  for (int s = 128; s > 0; s >>= 1) {
    if (threadIdx.x < s) red[threadIdx.x] += red[threadIdx.x + s];
    __syncthreads();
  }
  float mean = red[0] * (1.0f / D_);
  __syncthreads();
  float lv = 0.0f;
  for (int d = threadIdx.x; d < D_; d += 256) { float dd = xs[d] - mean; lv += dd * dd; }
  red[threadIdx.x] = lv; __syncthreads();
  for (int s = 128; s > 0; s >>= 1) {
    if (threadIdx.x < s) red[threadIdx.x] += red[threadIdx.x + s];
    __syncthreads();
  }
  float rstd = rsqrtf(red[0] * (1.0f / D_) + 1e-12f);
  for (int d = threadIdx.x; d < D_; d += 256)
    h[(long)t * D_ + d] = (__bf16)(g[d] * (xs[d] - mean) * rstd + bta[d]);
}

// ---------------------------------------------------------------------------
// 4) bf16 WMMA GEMM:  C[M,N] = act(A[M,K] @ B + bias),  B given as BT[N,K].
//    256 threads = 8 waves; block tile 128x128; wave tile 64x32 (4x2 WMMAs).
//    epi: 0 = bias only, 1 = bias + exact GELU.
// ---------------------------------------------------------------------------
__global__ __launch_bounds__(256) void gemm_bf16_wmma(
    const __bf16* __restrict__ A,
    const __bf16* __restrict__ BT,
    const float*  __restrict__ bias,
    __bf16* __restrict__ C,
    int M, int N, int K, int epi)
{
  const int lane   = threadIdx.x & 31;
  const int wave   = threadIdx.x >> 5;
  const int wm     = wave & 1;            // 2 waves along M
  const int wn     = wave >> 1;           // 4 waves along N
  const int laneLo = lane & 15;
  const int laneHi = lane >> 4;
  const int mBase  = blockIdx.y * 128 + wm * 64;
  const int nBase  = blockIdx.x * 128 + wn * 32;

  v8f acc[4][2];
  const v8f vzero = {0.f, 0.f, 0.f, 0.f, 0.f, 0.f, 0.f, 0.f};
#pragma unroll
  for (int i = 0; i < 4; ++i)
#pragma unroll
    for (int j = 0; j < 2; ++j) acc[i][j] = vzero;

  int mrow[4];
#pragma unroll
  for (int i = 0; i < 4; ++i) {
    int m = mBase + i * 16 + laneLo;
    mrow[i] = (m < M) ? m : (M - 1);      // clamp; stores are guarded
  }
  int ncol[2];
#pragma unroll
  for (int j = 0; j < 2; ++j) {
    int n = nBase + j * 16 + laneLo;
    ncol[j] = (n < N) ? n : (N - 1);
  }

  for (int k0 = 0; k0 < K; k0 += 32) {
    ABFrag a[4], b[2];
    // A 16x32 bf16 fragment: lanes 0-15 hold K 0..15, lanes 16-31 hold K 16..31
    const int kA = k0 + laneHi * 16;
#pragma unroll
    for (int i = 0; i < 4; ++i) {
      const __bf16* p = A + (long)mrow[i] * K + kA;
      a[i].h[0] = *(const v8bf*)(p);
      a[i].h[1] = *(const v8bf*)(p + 8);
    }
    // B 32x16 bf16 fragment from BT[N][K]: lane holds 16 contiguous K values
    const int kB = k0 + laneHi * 16;
#pragma unroll
    for (int j = 0; j < 2; ++j) {
      const __bf16* p = BT + (long)ncol[j] * K + kB;
      b[j].h[0] = *(const v8bf*)(p);
      b[j].h[1] = *(const v8bf*)(p + 8);
    }
#pragma unroll
    for (int i = 0; i < 4; ++i)
#pragma unroll
      for (int j = 0; j < 2; ++j)
        acc[i][j] = __builtin_amdgcn_wmma_f32_16x16x32_bf16(
            false, a[i].v, false, b[j].v, (short)0, acc[i][j], false, false);
  }

  // Epilogue: C/D layout — VGPR r: row = base + (lane>=16 ? 8 : 0) + r, col = base + lane%16
#pragma unroll
  for (int i = 0; i < 4; ++i) {
#pragma unroll
    for (int j = 0; j < 2; ++j) {
      int col = nBase + j * 16 + laneLo;
      if (col >= N) continue;
      float bv = bias[col];
#pragma unroll
      for (int r = 0; r < 8; ++r) {
        int row = mBase + i * 16 + laneHi * 8 + r;
        if (row >= M) continue;
        float v = acc[i][j][r] + bv;
        if (epi == 1) v = 0.5f * v * (1.0f + erff(v * 0.70710678118f));
        C[(long)row * N + col] = (__bf16)v;
      }
    }
  }
}

// ---------------------------------------------------------------------------
// 5) attention, fully on WMMA:
//    one block (256 thr, 8 waves) per (b, h, 16-q-row tile).
//    S = Q(16x64) @ K^T     -> WMMA, K rows read directly (already contiguous)
//    softmax in LDS (rows padded to 416 = 13*32, pad cols zeroed)
//    ctx = P(16x416) @ V    -> WMMA, V staged transposed in LDS per 32-kv chunk
// ---------------------------------------------------------------------------
__global__ __launch_bounds__(256) void attention_wmma_kernel(
    const __bf16* __restrict__ qkv,     // [M_, 3*D_]: Q | K | V
    const float* __restrict__ maskbias, // [B_, L_]
    __bf16* __restrict__ ctx)           // [M_, D_]
{
  const int qt = blockIdx.x;   // 0..24
  const int hh = blockIdx.y;   // 0..11
  const int bb = blockIdx.z;   // 0..15
  const int t  = threadIdx.x;
  const int lane   = t & 31;
  const int wave   = t >> 5;
  const int laneLo = lane & 15;
  const int laneHi = lane >> 4;
  const long rowbase = (long)bb * L_;

  __shared__ float  sc[16][LP_];     // scores -> probs (f32), 26.6 KB
  __shared__ __bf16 vt[HD_][40];     // V^T chunk [d][kv], stride 40 (80B, 16B-aligned)

  const v8f vzero = {0.f, 0.f, 0.f, 0.f, 0.f, 0.f, 0.f, 0.f};

  // zero the padded score columns (>= L_); never touched by the score WMMAs
  for (int e = t; e < 16 * LP_; e += 256) {
    int qi = e / LP_, c = e - qi * LP_;
    if (c >= L_) sc[qi][c] = 0.0f;
  }

  // ---- scores: S = Q @ K^T, distributed over waves by 16-col kv tile ----
  ABFrag aq[2];
  {
    int qi = qt * 16 + laneLo; if (qi >= L_) qi = L_ - 1;
    const __bf16* qp = qkv + (rowbase + qi) * (3 * D_) + hh * HD_;
#pragma unroll
    for (int kk = 0; kk < 2; ++kk) {
      const int kA = kk * 32 + laneHi * 16;
      aq[kk].h[0] = *(const v8bf*)(qp + kA);
      aq[kk].h[1] = *(const v8bf*)(qp + kA + 8);
    }
  }
  for (int nt = wave; nt < (L_ + 15) / 16; nt += 8) {
    v8f acc = vzero;
    int kv = nt * 16 + laneLo;
    int kvc = kv < L_ ? kv : L_ - 1;
    const __bf16* kp = qkv + (rowbase + kvc) * (3 * D_) + D_ + hh * HD_;
#pragma unroll
    for (int kk = 0; kk < 2; ++kk) {
      ABFrag bk;
      const int kB = kk * 32 + laneHi * 16;
      bk.h[0] = *(const v8bf*)(kp + kB);
      bk.h[1] = *(const v8bf*)(kp + kB + 8);
      acc = __builtin_amdgcn_wmma_f32_16x16x32_bf16(
          false, aq[kk].v, false, bk.v, (short)0, acc, false, false);
    }
    if (kv < L_) {
      float mb = maskbias[bb * L_ + kv];
#pragma unroll
      for (int r = 0; r < 8; ++r)
        sc[laneHi * 8 + r][kv] = acc[r] * 0.125f + mb;  // scale = 1/sqrt(64)
    }
  }
  __syncthreads();

  // ---- softmax per q row (16 serial rows; tiny) ----
  if (t < 16) {
    float mx = -3.4e38f;
    for (int kv = 0; kv < L_; ++kv) mx = fmaxf(mx, sc[t][kv]);
    float sum = 0.0f;
    for (int kv = 0; kv < L_; ++kv) { float e2 = __expf(sc[t][kv] - mx); sc[t][kv] = e2; sum += e2; }
    float inv = 1.0f / sum;
    for (int kv = 0; kv < L_; ++kv) sc[t][kv] *= inv;
  }
  __syncthreads();

  // ---- ctx = P @ V: 13 chunks of 32 kv; waves 0..3 own 16-wide d tiles ----
  v8f accc = vzero;
  for (int k0 = 0; k0 < LP_; k0 += 32) {
    // stage V^T chunk: vt[d][kvloc]; one v8bf global load per thread, coalesced in d
    {
      const int kvloc = t >> 3;           // 0..31
      const int dc    = t & 7;            // 8-elem d chunk
      const int kv    = k0 + kvloc;
      v8bf vv;
      if (kv < L_) {
        vv = *(const v8bf*)(qkv + (rowbase + kv) * (3 * D_) + 2 * D_ + hh * HD_ + dc * 8);
      } else {
#pragma unroll
        for (int i = 0; i < 8; ++i) vv[i] = (__bf16)0.0f;
      }
#pragma unroll
      for (int i = 0; i < 8; ++i) vt[dc * 8 + i][kvloc] = vv[i];
    }
    __syncthreads();
    if (wave < 4) {
      // A fragment from prob rows in LDS (f32 -> bf16)
      ABFrag ap;
      {
        const int qi = laneLo;
        const int kbase = k0 + laneHi * 16;
#pragma unroll
        for (int i = 0; i < 8; ++i) ap.v[i] = (__bf16)sc[qi][kbase + i];
#pragma unroll
        for (int i = 0; i < 8; ++i) ap.v[i + 8] = (__bf16)sc[qi][kbase + 8 + i];
      }
      // B fragment from transposed V chunk in LDS (16 contiguous kv per lane)
      ABFrag bv;
      {
        const __bf16* p = &vt[wave * 16 + laneLo][laneHi * 16];
        bv.h[0] = *(const v8bf*)(p);
        bv.h[1] = *(const v8bf*)(p + 8);
      }
      accc = __builtin_amdgcn_wmma_f32_16x16x32_bf16(
          false, ap.v, false, bv.v, (short)0, accc, false, false);
    }
    __syncthreads();
  }
  if (wave < 4) {
    const int col = hh * HD_ + wave * 16 + laneLo;
#pragma unroll
    for (int r = 0; r < 8; ++r) {
      int row = qt * 16 + laneHi * 8 + r;
      if (row < L_) ctx[(rowbase + row) * D_ + col] = (__bf16)accc[r];
    }
  }
}

// ---------------------------------------------------------------------------
// 6) residual add + LayerNorm (in-place on h)
// ---------------------------------------------------------------------------
__global__ __launch_bounds__(256) void add_ln_kernel(
    const __bf16* __restrict__ hin, const __bf16* __restrict__ delta,
    const float* __restrict__ g, const float* __restrict__ bta,
    __bf16* __restrict__ hout)
{
  const int t = blockIdx.x;
  __shared__ float xs[D_];
  __shared__ float red[256];
  float local = 0.0f;
  for (int d = threadIdx.x; d < D_; d += 256) {
    float x = (float)hin[(long)t * D_ + d] + (float)delta[(long)t * D_ + d];
    xs[d] = x; local += x;
  }
  red[threadIdx.x] = local; __syncthreads();
  for (int s = 128; s > 0; s >>= 1) {
    if (threadIdx.x < s) red[threadIdx.x] += red[threadIdx.x + s];
    __syncthreads();
  }
  float mean = red[0] * (1.0f / D_);
  __syncthreads();
  float lv = 0.0f;
  for (int d = threadIdx.x; d < D_; d += 256) { float dd = xs[d] - mean; lv += dd * dd; }
  red[threadIdx.x] = lv; __syncthreads();
  for (int s = 128; s > 0; s >>= 1) {
    if (threadIdx.x < s) red[threadIdx.x] += red[threadIdx.x + s];
    __syncthreads();
  }
  float rstd = rsqrtf(red[0] * (1.0f / D_) + 1e-12f);
  for (int d = threadIdx.x; d < D_; d += 256)
    hout[(long)t * D_ + d] = (__bf16)(g[d] * (xs[d] - mean) * rstd + bta[d]);
}

// ---------------------------------------------------------------------------
// 7) pooler: tanh(h[:,0] @ pool_W + pool_b)   (M=16 — VALU is fine here)
// ---------------------------------------------------------------------------
__global__ __launch_bounds__(256) void pool_kernel(
    const __bf16* __restrict__ h, const float* __restrict__ pW,
    const float* __restrict__ pb, float* __restrict__ pool)
{
  const int bb = blockIdx.x;
  const __bf16* hr = h + (long)(bb * L_) * D_;   // [CLS] row
  for (int j = threadIdx.x; j < D_; j += 256) {
    float s = pb[j];
    for (int k = 0; k < D_; ++k) s += (float)hr[k] * pW[(long)k * D_ + j];
    pool[bb * D_ + j] = tanhf(s);
  }
}

__global__ void logits_kernel(const float* __restrict__ pool,
                              const float* __restrict__ lW,
                              const float* __restrict__ lb,
                              float* __restrict__ out)
{
  int t = threadIdx.x;
  if (t >= B_ * 2) return;
  int bb = t >> 1, o = t & 1;
  float s = lb[o];
  for (int k = 0; k < D_; ++k) s += pool[bb * D_ + k] * lW[k * 2 + o];
  out[bb * 2 + o] = s;
}

// ---------------------------------------------------------------------------
// host launcher
// ---------------------------------------------------------------------------
extern "C" void kernel_launch(void* const* d_in, const int* in_sizes, int n_in,
                              void* d_out, int out_size, void* d_ws, size_t ws_size,
                              hipStream_t stream) {
  (void)in_sizes; (void)n_in; (void)out_size; (void)ws_size;

  const int*   inpids     = (const int*)  d_in[0];
  const int*   map_table  = (const int*)  d_in[1];
  const int*   map_counts = (const int*)  d_in[2];
  const float* word_emb   = (const float*)d_in[3];
  const float* pos_emb    = (const float*)d_in[4];
  const float* type_emb   = (const float*)d_in[5];
  const float* emb_ln_g   = (const float*)d_in[6];
  const float* emb_ln_b   = (const float*)d_in[7];
  const float* Wqkv       = (const float*)d_in[8];
  const float* bqkv       = (const float*)d_in[9];
  const float* Wo         = (const float*)d_in[10];
  const float* bo         = (const float*)d_in[11];
  const float* ln1_g      = (const float*)d_in[12];
  const float* ln1_b      = (const float*)d_in[13];
  const float* W1         = (const float*)d_in[14];
  const float* b1         = (const float*)d_in[15];
  const float* W2         = (const float*)d_in[16];
  const float* b2         = (const float*)d_in[17];
  const float* ln2_g      = (const float*)d_in[18];
  const float* ln2_b      = (const float*)d_in[19];
  const float* pool_W     = (const float*)d_in[20];
  const float* pool_b     = (const float*)d_in[21];
  const float* lin_W      = (const float*)d_in[22];
  const float* lin_b      = (const float*)d_in[23];
  float* out = (float*)d_out;

  // ---- workspace carve-out (aligned 256B) ----
  char* base = (char*)d_ws;
  size_t off = 0;
  auto alloc = [&](size_t bytes) -> char* {
    char* p = base + off;
    off = (off + bytes + 255) & ~(size_t)255;
    return p;
  };
  int*     ws_newinp = (int*)   alloc((size_t)M_ * 4);
  float*   ws_mask   = (float*) alloc((size_t)M_ * 4);
  __bf16*  ws_h      = (__bf16*)alloc((size_t)M_ * D_ * 2);
  __bf16*  ws_qkv    = (__bf16*)alloc((size_t)M_ * 3 * D_ * 2);
  __bf16*  ws_ctx    = (__bf16*)alloc((size_t)M_ * D_ * 2);
  __bf16*  ws_gout   = (__bf16*)alloc((size_t)M_ * D_ * 2);
  __bf16*  ws_ff1    = (__bf16*)alloc((size_t)M_ * FF_ * 2);
  float*   ws_pool   = (float*) alloc((size_t)B_ * D_ * 4);
  __bf16*  ws_WqkvT  = (__bf16*)alloc((size_t)NL_ * D_ * 3 * D_ * 2);
  __bf16*  ws_WoT    = (__bf16*)alloc((size_t)NL_ * D_ * D_ * 2);
  __bf16*  ws_W1T    = (__bf16*)alloc((size_t)NL_ * D_ * FF_ * 2);
  __bf16*  ws_W2T    = (__bf16*)alloc((size_t)NL_ * FF_ * D_ * 2);

  // 1) remap + mask
  remap_kernel<<<1, 32, 0, stream>>>(inpids, map_table, map_counts, ws_newinp, ws_mask);

  // 2) weight convert + transpose to bf16
  {
    long t;
    t = (long)NL_ * D_ * 3 * D_;
    convt_kernel<<<(unsigned)((t + 255) / 256), 256, 0, stream>>>(Wqkv, ws_WqkvT, D_, 3 * D_, t);
    t = (long)NL_ * D_ * D_;
    convt_kernel<<<(unsigned)((t + 255) / 256), 256, 0, stream>>>(Wo, ws_WoT, D_, D_, t);
    t = (long)NL_ * D_ * FF_;
    convt_kernel<<<(unsigned)((t + 255) / 256), 256, 0, stream>>>(W1, ws_W1T, D_, FF_, t);
    t = (long)NL_ * FF_ * D_;
    convt_kernel<<<(unsigned)((t + 255) / 256), 256, 0, stream>>>(W2, ws_W2T, FF_, D_, t);
  }

  // 3) embeddings + LN
  embed_ln_kernel<<<M_, 256, 0, stream>>>(ws_newinp, word_emb, pos_emb, type_emb,
                                          emb_ln_g, emb_ln_b, ws_h);

  const int gridM = (M_ + 127) / 128;   // 49
  // 4) encoder layers
  for (int l = 0; l < NL_; ++l) {
    // qkv = h @ Wqkv + bqkv
    gemm_bf16_wmma<<<dim3((3 * D_) / 128, gridM), 256, 0, stream>>>(
        ws_h, ws_WqkvT + (size_t)l * D_ * 3 * D_, bqkv + (size_t)l * 3 * D_,
        ws_qkv, M_, 3 * D_, D_, 0);
    // attention (WMMA)
    attention_wmma_kernel<<<dim3((L_ + 15) / 16, H_, B_), 256, 0, stream>>>(
        ws_qkv, ws_mask, ws_ctx);
    // proj = ctx @ Wo + bo
    gemm_bf16_wmma<<<dim3(D_ / 128, gridM), 256, 0, stream>>>(
        ws_ctx, ws_WoT + (size_t)l * D_ * D_, bo + (size_t)l * D_,
        ws_gout, M_, D_, D_, 0);
    // h = LN(h + proj)
    add_ln_kernel<<<M_, 256, 0, stream>>>(ws_h, ws_gout,
                                          ln1_g + (size_t)l * D_, ln1_b + (size_t)l * D_, ws_h);
    // ff1 = gelu(h @ W1 + b1)
    gemm_bf16_wmma<<<dim3(FF_ / 128, gridM), 256, 0, stream>>>(
        ws_h, ws_W1T + (size_t)l * D_ * FF_, b1 + (size_t)l * FF_,
        ws_ff1, M_, FF_, D_, 1);
    // ff2 = ff1 @ W2 + b2
    gemm_bf16_wmma<<<dim3(D_ / 128, gridM), 256, 0, stream>>>(
        ws_ff1, ws_W2T + (size_t)l * FF_ * D_, b2 + (size_t)l * D_,
        ws_gout, M_, D_, FF_, 0);
    // h = LN(h + ff2)
    add_ln_kernel<<<M_, 256, 0, stream>>>(ws_h, ws_gout,
                                          ln2_g + (size_t)l * D_, ln2_b + (size_t)l * D_, ws_h);
  }

  // 5) pooler + classifier
  pool_kernel<<<B_, 256, 0, stream>>>(ws_h, pool_W, pool_b, ws_pool);
  logits_kernel<<<1, 32, 0, stream>>>(ws_pool, lin_W, lin_b, out);
}
